// PointerSelectorV4_7_46780783788511
// MI455X (gfx1250) — matrix operand order repaired and checked by reading, not compiled, not run on hardware
//
#include <hip/hip_runtime.h>
#include <stdint.h>

typedef float  v8f  __attribute__((ext_vector_type(8)));
typedef __bf16 v16bf __attribute__((ext_vector_type(16)));
typedef __bf16 v8bf  __attribute__((ext_vector_type(8)));
typedef __bf16 v2bf  __attribute__((ext_vector_type(2)));

static constexpr int B_ = 8192, K_ = 32, D_ = 768, H_ = 256, R_ = 32, S_ = 4;

// ---------------- workspace layout (bytes) ----------------
static constexpr size_t OFF_CBF  = 0;                          // B*K x H   bf16 (c projection)
static constexpr size_t SZ_CBF   = (size_t)B_ * K_ * H_ * 2;
static constexpr size_t OFF_Q    = OFF_CBF + SZ_CBF;           // B x H     f32  (q projection)
static constexpr size_t SZ_Q     = (size_t)B_ * H_ * 4;
static constexpr size_t OFF_PWIN = OFF_Q + SZ_Q;               // packed W_in   (K=768,N=256)
static constexpr size_t SZ_PWIN  = (size_t)D_ * H_ * 2;
static constexpr size_t SZ_P256  = (size_t)H_ * H_ * 2;
static constexpr size_t OFF_PKP  = OFF_PWIN + SZ_PWIN;         // packed Wkp (Wvp, Wc follow contiguously)
static constexpr size_t OFF_PVP  = OFF_PKP + SZ_P256;
static constexpr size_t OFF_PWC  = OFF_PVP + SZ_P256;
static constexpr size_t OFF_LOSS = OFF_PWC + SZ_P256;          // B*S f32 loss terms

static constexpr size_t SMEM_GEMM  = 128*40*2 + 128*256*2;     // A-stage + out-stage
static constexpr size_t SMEM_FUSED = 132608;                   // carve below (~131972 B)

// ---------------------------------------------------------------------------
// Pack a row-major weight W (N x K, f32) into a contiguous-k B-fragment layout for
// v_wmma_f32_16x16x32_bf16:  dword index = ((tn*KT + tk)*32 + lane)*8 + j
// lane: half=lane>>4 selects K half-chunk, col=lane&15 selects N; dword j
// holds bf16 pair {W[n][k0+2j], W[n][k0+2j+1]} with k0 = tk*32 + 16*half.
// ---------------------------------------------------------------------------
__global__ void pack_b_kernel(const float* __restrict__ W, uint32_t* __restrict__ dst,
                              int K, int total) {
    int tid = blockIdx.x * 256 + threadIdx.x;
    if (tid >= total) return;
    int j    = tid & 7;
    int lane = (tid >> 3) & 31;
    int tile = tid >> 8;
    int KT   = K >> 5;
    int tk = tile % KT, tn = tile / KT;
    int half = lane >> 4, cl = lane & 15;
    int n = tn * 16 + cl;
    int k = tk * 32 + half * 16 + 2 * j;
    v2bf p;
    p.x = (__bf16)W[(size_t)n * K + k];
    p.y = (__bf16)W[(size_t)n * K + k + 1];
    dst[tid] = __builtin_bit_cast(uint32_t, p);
}

// ---------------------------------------------------------------------------
// WMMA GEMM: out[M x 256] = A[M x 768] @ (packed W_in)^T.  128 rows / block,
// wave w owns rows [w*16, w*16+16) and all 16 N-tiles (128 accum VGPRs).
// A staged to LDS as bf16 (row stride 40 elements: 16B-aligned, bank-spread).
// ---------------------------------------------------------------------------
template <bool OUT_BF16>
__global__ __launch_bounds__(256) void gemm_win_wmma(const float* __restrict__ A,
                                                     const uint32_t* __restrict__ Bp,
                                                     float* __restrict__ outF,
                                                     uint16_t* __restrict__ outB16) {
    extern __shared__ char smem[];
    __bf16* Abf = (__bf16*)smem;                     // [128][40]
    __bf16* ost = (__bf16*)(smem + 128 * 40 * 2);    // [128][256]
    const int tid  = threadIdx.x;
    const int lane = tid & 31, wv = tid >> 5;
    const int half = lane >> 4, lo = lane & 15;
    const size_t m0 = (size_t)blockIdx.x * 128;

    const v8f vzero = {0.f, 0.f, 0.f, 0.f, 0.f, 0.f, 0.f, 0.f};
    v8f acc[16];
#pragma unroll
    for (int i = 0; i < 16; ++i) acc[i] = vzero;

    const int rowA = wv * 16 + lo;
    const int srow = tid >> 1, sseg = tid & 1;
    const float* gbase = A + (m0 + srow) * D_ + sseg * 16;
    for (int tk = 0; tk < D_ / 32; ++tk) {
        const int k0 = tk * 32;
        {   // stage A chunk [128][32] as bf16
            const float* g = gbase + k0;
            float4 f0 = ((const float4*)g)[0];
            float4 f1 = ((const float4*)g)[1];
            float4 f2 = ((const float4*)g)[2];
            float4 f3 = ((const float4*)g)[3];
            __builtin_prefetch(g + 32, 0, 1);        // next K-chunk -> global_prefetch
            v8bf b0, b1;
            b0[0]=(__bf16)f0.x; b0[1]=(__bf16)f0.y; b0[2]=(__bf16)f0.z; b0[3]=(__bf16)f0.w;
            b0[4]=(__bf16)f1.x; b0[5]=(__bf16)f1.y; b0[6]=(__bf16)f1.z; b0[7]=(__bf16)f1.w;
            b1[0]=(__bf16)f2.x; b1[1]=(__bf16)f2.y; b1[2]=(__bf16)f2.z; b1[3]=(__bf16)f2.w;
            b1[4]=(__bf16)f3.x; b1[5]=(__bf16)f3.y; b1[6]=(__bf16)f3.z; b1[7]=(__bf16)f3.w;
            __bf16* d = Abf + srow * 40 + sseg * 16;
            *(v8bf*)d       = b0;
            *(v8bf*)(d + 8) = b1;
        }
        __syncthreads();
        // A fragment: 16 contiguous K per lane (k = 16*half + s, matches packed B)
        const __bf16* ap = Abf + rowA * 40 + 16 * half;
        v8bf alo = *(const v8bf*)ap;
        v8bf ahi = *(const v8bf*)(ap + 8);
        v16bf afrag;
#pragma unroll
        for (int i = 0; i < 8; ++i) { afrag[i] = alo[i]; afrag[i + 8] = ahi[i]; }
#pragma unroll
        for (int nt = 0; nt < 16; ++nt) {
            v16bf bfrag = *(const v16bf*)(Bp + (((size_t)(nt * (D_ / 32) + tk)) * 32 + lane) * 8);
            acc[nt] = __builtin_amdgcn_wmma_f32_16x16x32_bf16(
                false, afrag, false, bfrag, (short)0, acc[nt], false, false);
        }
        __syncthreads();
    }
    if constexpr (OUT_BF16) {
#pragma unroll
        for (int nt = 0; nt < 16; ++nt)
#pragma unroll
            for (int r = 0; r < 8; ++r)
                ost[(wv * 16 + r + 8 * half) * 256 + nt * 16 + lo] = (__bf16)acc[nt][r];
        __syncthreads();
        for (int c = tid; c < 4096; c += 256) {   // coalesced 16B-chunk stores
            int row = c >> 5, off = (c & 31) * 8;
            *(v8bf*)(((__bf16*)outB16) + (m0 + row) * 256 + off) =
                *(const v8bf*)(ost + row * 256 + off);
        }
    } else {
#pragma unroll
        for (int nt = 0; nt < 16; ++nt)
#pragma unroll
            for (int r = 0; r < 8; ++r)
                outF[(m0 + wv * 16 + r + 8 * half) * 256 + nt * 16 + lo] = acc[nt][r];
    }
}

// ---------------------------------------------------------------------------
__device__ __forceinline__ float dot256(const float* a, const float* b) {
    float s0 = 0.f, s1 = 0.f;
#pragma unroll 4
    for (int i = 0; i < 256; i += 2) {
        float2 x = *(const float2*)(a + i);
        float2 y = *(const float2*)(b + i);
        s0 = fmaf(x.x, y.x, s0);
        s1 = fmaf(x.y, y.y, s1);
    }
    return s0 + s1;
}

__device__ __forceinline__ float block_sum(float v, float* red) {
    int t = threadIdx.x;
    red[t] = v;
    __syncthreads();
    for (int s = 128; s > 0; s >>= 1) {
        if (t < s) red[t] += red[t + s];
        __syncthreads();
    }
    float r = red[0];
    __syncthreads();
    return r;
}

// ---------------------------------------------------------------------------
// One block per batch element b: attention encode + DPP + 4-step GRU selector.
// kh/vh/cand_n computed with in-block WMMA from the LDS-resident bf16 c_b.
// The three packed weights are contiguous, so Bp / dst / bias are computed
// arithmetically from the job's g -> uniform global_load addressing (no FLAT).
// ---------------------------------------------------------------------------
__global__ __launch_bounds__(256) void fused_selector(
    const uint16_t* __restrict__ cbf_g, const float* __restrict__ q_g,
    const uint32_t* __restrict__ pKVC,    // packed Wkp|Wvp|Wc, contiguous
    const float* __restrict__ attn_in_w, const float* __restrict__ attn_in_b,
    const float* __restrict__ attn_out_w, const float* __restrict__ attn_out_b,
    const float* __restrict__ ln_g, const float* __restrict__ ln_b,
    const float* __restrict__ Wq,
    const float* __restrict__ gru_wih, const float* __restrict__ gru_whh,
    const float* __restrict__ gru_bih, const float* __restrict__ gru_bhh,
    const float* __restrict__ step_emb, const float* __restrict__ W_dpp,
    const float* __restrict__ dpp_lambda, const int* __restrict__ labels,
    float* __restrict__ logits_out, float* __restrict__ preds_out,
    float* __restrict__ loss_out) {
    extern __shared__ char smem[];
    __bf16* cbf = (__bf16*)smem;                       // [32][264] bf16
    float* kh   = (float*)(smem + 32 * 264 * 2);       // [32][258]  (vh, cn contiguous after)
    float* vh   = kh + 32 * 258;
    float* cn   = vh + 32 * 258;
    float* dpp  = cn + 32 * 258;                       // [32][33]
    float* qv   = dpp + 32 * 33;
    float* qh   = qv + 256;
    float* tv   = qh + 256;
    float* xv   = tv + 256;
    float* hv   = xv + 256;
    float* hev  = hv + 256;
    float* red  = hev + 256;
    float* bq   = red + 256;
    float* bias3 = bq + 256;                           // [768]: bk | bv | zeros
    float* att  = bias3 + 768;                         // [32]
    float* score = att + 32;
    float* msim  = score + 32;
    int*   mask  = (int*)(msim + 32);                  // [32]
    int*   seli  = mask + 32;

    const int b = blockIdx.x;
    const int tid = threadIdx.x, lane = tid & 31, wv = tid >> 5;
    const int half = lane >> 4, lo = lane & 15;

    // ---- P0: stage c_b (bf16), q_b, biases ----
    {
        const uint32_t* src = (const uint32_t*)(cbf_g + (size_t)b * K_ * H_);
        for (int i = tid; i < K_ * H_ / 2; i += 256) {
            int row = i >> 7, cp = (i & 127) * 2;
            *(uint32_t*)(cbf + row * 264 + cp) = src[i];
        }
        qv[tid]  = q_g[(size_t)b * H_ + tid];
        bq[tid]  = attn_in_b[tid];
        bias3[tid]       = attn_in_b[H_ + tid];
        bias3[256 + tid] = attn_in_b[2 * H_ + tid];
        bias3[512 + tid] = 0.f;
        if (tid < 32) { msim[tid] = -INFINITY; mask[tid] = 0; }
    }
    __syncthreads();

    // ---- P1: qh = q @ Wqp^T + bq (matvec, VALU) ----
    qh[tid] = dot256(qv, attn_in_w + (size_t)tid * H_) + bq[tid];

    // ---- P2: WMMA  kh = c@Wkp^T+bk,  vh = c@Wvp^T+bv,  cn_raw = c@Wc^T ----
    for (int job = wv; job < 96; job += 8) {
        int g = job >> 5, t = job & 31;
        int mt = t & 1, nt = t >> 1;
        const uint32_t* Bp = pKVC + (size_t)g * (H_ * H_ / 2);
        const v8f vzero = {0.f, 0.f, 0.f, 0.f, 0.f, 0.f, 0.f, 0.f};
        v8f acc = vzero;
        const __bf16* abase = cbf + (mt * 16 + lo) * 264 + 16 * half;
#pragma unroll
        for (int ks = 0; ks < 8; ++ks) {
            const __bf16* ap = abase + ks * 32;
            v8bf alo = *(const v8bf*)ap;
            v8bf ahi = *(const v8bf*)(ap + 8);
            v16bf afrag;
#pragma unroll
            for (int i = 0; i < 8; ++i) { afrag[i] = alo[i]; afrag[i + 8] = ahi[i]; }
            v16bf bfrag = *(const v16bf*)(Bp + (((size_t)(nt * 8 + ks)) * 32 + lane) * 8);
            acc = __builtin_amdgcn_wmma_f32_16x16x32_bf16(
                false, afrag, false, bfrag, (short)0, acc, false, false);
        }
        float* dst = kh + (size_t)g * (32 * 258);      // kh / vh / cn contiguous
        const float* bsel = bias3 + g * 256;           // bk / bv / zeros
#pragma unroll
        for (int r = 0; r < 8; ++r) {
            int row = mt * 16 + r + 8 * half;
            int cg  = nt * 16 + lo;
            dst[row * 258 + cg] = acc[r] + bsel[cg];
        }
    }
    __syncthreads();

    // ---- P3: cand_n = l2norm rows of cn ----
    if (tid < 32) {
        float ss = 0.f;
        const float* row = cn + tid * 258;
        for (int i = 0; i < 256; ++i) ss = fmaf(row[i], row[i], ss);
        red[tid] = 1.0f / fmaxf(sqrtf(ss), 1e-12f);
    }
    __syncthreads();
    {
        int m = tid >> 3, c0 = (tid & 7) * 32;
        float inv = red[m];
        float* row = cn + m * 258 + c0;
        for (int i = 0; i < 32; ++i) row[i] *= inv;
    }
    __syncthreads();

    // ---- P4: attention softmax ----
    if (tid < 32) score[tid] = dot256(qh, kh + tid * 258) * 0.0625f;  // 1/sqrt(H)
    __syncthreads();
    if (tid == 0) {
        float mx = score[0];
        for (int k = 1; k < 32; ++k) mx = fmaxf(mx, score[k]);
        float se = 0.f;
        for (int k = 0; k < 32; ++k) { att[k] = expf(score[k] - mx); se += att[k]; }
        float inv = 1.0f / se;
        for (int k = 0; k < 32; ++k) att[k] *= inv;
    }
    __syncthreads();

    // ---- P5: ao, out-proj, residual ----
    {
        float s = 0.f;
        for (int k = 0; k < 32; ++k) s = fmaf(att[k], vh[k * 258 + tid], s);
        tv[tid] = s;
    }
    __syncthreads();
    xv[tid] = dot256(tv, attn_out_w + (size_t)tid * H_) + attn_out_b[tid] + qv[tid];
    __syncthreads();

    // ---- P6: LayerNorm ----
    {
        float xs = xv[tid];
        float s1 = block_sum(xs, red);
        float s2 = block_sum(xs * xs, red);
        float mu = s1 * (1.0f / 256.0f);
        float var = s2 * (1.0f / 256.0f) - mu * mu;
        xv[tid] = (xs - mu) * rsqrtf(var + 1e-5f) * ln_g[tid] + ln_b[tid];
    }
    __syncthreads();

    // ---- P7: h = l2norm(query_enh @ Wq^T) ----
    {
        float th = dot256(xv, Wq + (size_t)tid * H_);
        float ss = block_sum(th * th, red);
        hv[tid] = th * (1.0f / fmaxf(sqrtf(ss), 1e-12f));
    }
    __syncthreads();

    // ---- P8: dpp = l2norm(cand_n @ W_dpp^T) ----
    for (int it = 0; it < 4; ++it) {
        int idx = tid + it * 256;
        int m = idx >> 5, r = idx & 31;
        dpp[m * 33 + r] = dot256(cn + m * 258, W_dpp + (size_t)r * H_);
    }
    __syncthreads();
    if (tid < 32) {
        float ss = 0.f;
        for (int r = 0; r < 32; ++r) { float v = dpp[tid * 33 + r]; ss = fmaf(v, v, ss); }
        red[tid] = 1.0f / fmaxf(sqrtf(ss), 1e-12f);
    }
    __syncthreads();
    for (int it = 0; it < 4; ++it) {
        int idx = tid + it * 256;
        int m = idx >> 5, r = idx & 31;
        dpp[m * 33 + r] *= red[m];
    }
    __syncthreads();

    const float lam = log1pf(expf(dpp_lambda[0]));  // softplus

    // ---- P10: 4-step pointer/GRU loop ----
    for (int s = 0; s < S_; ++s) {
        float hval = hv[tid] + step_emb[s * H_ + tid];   // s < MAX_SHOT
        float sse = block_sum(hval * hval, red);
        hev[tid] = hval * (1.0f / fmaxf(sqrtf(sse), 1e-12f));
        __syncthreads();
        if (tid < 32) {
            float base = dot256(hev, cn + tid * 258) * 10.0f;  // 1/TEMP
            float sc = base;
            if (s > 0) {
                float ms = msim[tid];
                sc = base + lam * logf(1e-6f + 1.0f - fminf(ms * ms, 0.999f));
            }
            if (mask[tid]) sc = -100.0f;
            score[tid] = sc;
            logits_out[((size_t)b * S_ + s) * K_ + tid] = sc;
        }
        __syncthreads();
        if (tid == 0) {
            int idx = labels[b * S_ + s];
            *seli = idx;
            float mx = score[0]; int am = 0;
            for (int k = 1; k < 32; ++k) if (score[k] > mx) { mx = score[k]; am = k; }
            preds_out[b * S_ + s] = (float)am;
            float se = 0.f, ssum = 0.f;
            for (int k = 0; k < 32; ++k) { se += expf(score[k] - mx); ssum += score[k]; }
            float logZ = mx + logf(se);
            float nll = logZ - score[idx];
            float smooth = logZ - ssum * (1.0f / 32.0f);
            loss_out[b * S_ + s] = 0.9f * nll + 0.1f * smooth;
            mask[idx] = 1;
        }
        __syncthreads();
        int sel = *seli;
        if (tid < 32) {  // max_sim update with chosen dpp row
            float dv = 0.f;
            const float* dk = dpp + tid * 33;
            const float* dc = dpp + sel * 33;
            for (int r = 0; r < 32; ++r) dv = fmaf(dk[r], dc[r], dv);
            msim[tid] = fmaxf(msim[tid], dv);
        }
        // GRU update (thread tid owns hidden channel tid)
        const float* ch = cn + sel * 258;
        float gir = dot256(ch, gru_wih + (size_t)tid * H_)            + gru_bih[tid];
        float giz = dot256(ch, gru_wih + (size_t)(H_ + tid) * H_)     + gru_bih[H_ + tid];
        float gin = dot256(ch, gru_wih + (size_t)(2 * H_ + tid) * H_) + gru_bih[2 * H_ + tid];
        float ghr = dot256(hv, gru_whh + (size_t)tid * H_)            + gru_bhh[tid];
        float ghz = dot256(hv, gru_whh + (size_t)(H_ + tid) * H_)     + gru_bhh[H_ + tid];
        float ghn = dot256(hv, gru_whh + (size_t)(2 * H_ + tid) * H_) + gru_bhh[2 * H_ + tid];
        float r_ = 1.0f / (1.0f + expf(-(gir + ghr)));
        float z_ = 1.0f / (1.0f + expf(-(giz + ghz)));
        float n_ = tanhf(gin + r_ * ghn);
        float hn = (1.0f - z_) * n_ + z_ * hv[tid];
        float ss2 = block_sum(hn * hn, red);   // also fences all hv reads
        hv[tid] = hn * (1.0f / fmaxf(sqrtf(ss2), 1e-12f));
        __syncthreads();
    }
}

// ---------------------------------------------------------------------------
__global__ void loss_reduce(const float* __restrict__ lossv, float* __restrict__ out) {
    __shared__ float red[256];
    float s = 0.f;
    for (int i = threadIdx.x; i < B_ * S_; i += 256) s += lossv[i];
    red[threadIdx.x] = s;
    __syncthreads();
    for (int k = 128; k > 0; k >>= 1) {
        if (threadIdx.x < k) red[threadIdx.x] += red[threadIdx.x + k];
        __syncthreads();
    }
    if (threadIdx.x == 0) out[0] = red[0] * (1.0f / (B_ * S_));
}

// ---------------------------------------------------------------------------
extern "C" void kernel_launch(void* const* d_in, const int* in_sizes, int n_in,
                              void* d_out, int out_size, void* d_ws, size_t ws_size,
                              hipStream_t stream) {
    (void)in_sizes; (void)n_in; (void)out_size; (void)ws_size;
    const float* query_emb  = (const float*)d_in[0];
    const float* cand_emb   = (const float*)d_in[1];
    const int*   labels     = (const int*)d_in[2];
    const float* W_in       = (const float*)d_in[3];
    const float* attn_in_w  = (const float*)d_in[4];
    const float* attn_in_b  = (const float*)d_in[5];
    const float* attn_out_w = (const float*)d_in[6];
    const float* attn_out_b = (const float*)d_in[7];
    const float* ln_g       = (const float*)d_in[8];
    const float* ln_b       = (const float*)d_in[9];
    const float* Wq         = (const float*)d_in[10];
    const float* Wc         = (const float*)d_in[11];
    const float* gru_wih    = (const float*)d_in[12];
    const float* gru_whh    = (const float*)d_in[13];
    const float* gru_bih    = (const float*)d_in[14];
    const float* gru_bhh    = (const float*)d_in[15];
    const float* step_emb   = (const float*)d_in[16];
    const float* W_dpp      = (const float*)d_in[17];
    const float* dpp_lambda = (const float*)d_in[18];

    uint8_t*  ws    = (uint8_t*)d_ws;
    uint16_t* cbf   = (uint16_t*)(ws + OFF_CBF);
    float*    qproj = (float*)(ws + OFF_Q);
    uint32_t* pWin  = (uint32_t*)(ws + OFF_PWIN);
    uint32_t* pKp   = (uint32_t*)(ws + OFF_PKP);
    uint32_t* pVp   = (uint32_t*)(ws + OFF_PVP);
    uint32_t* pWc   = (uint32_t*)(ws + OFF_PWC);
    float*    lossw = (float*)(ws + OFF_LOSS);

    float* logits = (float*)d_out;
    float* preds  = logits + (size_t)B_ * S_ * K_;
    float* lossf  = preds + (size_t)B_ * S_;

    // 1) pack weights into WMMA B-fragment layout (Wkp|Wvp|Wc contiguous)
    pack_b_kernel<<<(D_ * H_ / 2) / 256, 256, 0, stream>>>(W_in, pWin, D_, D_ * H_ / 2);
    pack_b_kernel<<<(H_ * H_ / 2) / 256, 256, 0, stream>>>(attn_in_w + H_ * H_, pKp, H_, H_ * H_ / 2);
    pack_b_kernel<<<(H_ * H_ / 2) / 256, 256, 0, stream>>>(attn_in_w + 2 * H_ * H_, pVp, H_, H_ * H_ / 2);
    pack_b_kernel<<<(H_ * H_ / 2) / 256, 256, 0, stream>>>(Wc, pWc, H_, H_ * H_ / 2);

    // 2) big WMMA GEMMs: c (bf16 out) and q (f32 out)
    gemm_win_wmma<true ><<<(B_ * K_) / 128, 256, SMEM_GEMM, stream>>>(cand_emb, pWin, nullptr, cbf);
    gemm_win_wmma<false><<<B_ / 128, 256, SMEM_GEMM, stream>>>(query_emb, pWin, qproj, nullptr);

    // 3) fused per-batch selector (in-block WMMA for kh/vh/cand_n)
    fused_selector<<<B_, 256, SMEM_FUSED, stream>>>(
        cbf, qproj, pKp, attn_in_w, attn_in_b, attn_out_w, attn_out_b,
        ln_g, ln_b, Wq, gru_wih, gru_whh, gru_bih, gru_bhh, step_emb, W_dpp,
        dpp_lambda, labels, logits, preds, lossw);

    // 4) mean loss
    loss_reduce<<<1, 256, 0, stream>>>(lossw, lossf);
}
